// RelationalInferer_48498770706570
// MI455X (gfx1250) — matrix operand order repaired and checked by reading, not compiled, not run on hardware
//
#include <hip/hip_runtime.h>
#include <math.h>

// ---------------------------------------------------------------------------
// Problem constants (from the reference)
// ---------------------------------------------------------------------------
#define B_   8
#define NC_  1024
#define NO_  4096
#define DIN_ 128
#define DK_  64
#define DV_  64
#define DH_  256
#define DP_  32
#define MQ_  (B_ * NC_)   // 8192 query rows
#define MG_  (B_ * NO_)   // 32768 object rows

typedef __attribute__((ext_vector_type(16))) _Float16 v16h;
typedef __attribute__((ext_vector_type(8)))  _Float16 v8h;
typedef __attribute__((ext_vector_type(8)))  float    v8f;

#define DEV static __device__ __forceinline__

DEV v8f zero8() {
  v8f z;
#pragma unroll
  for (int i = 0; i < 8; ++i) z[i] = 0.0f;
  return z;
}

DEV v16h join8(v8h lo, v8h hi) {
  v16h r;
#pragma unroll
  for (int i = 0; i < 8; ++i) { r[i] = lo[i]; r[8 + i] = hi[i]; }
  return r;
}

DEV v8f wmma_f16(v16h a, v16h b, v8f c) {
  return __builtin_amdgcn_wmma_f32_16x16x32_f16(
      /*neg_a=*/false, a, /*neg_b=*/false, b,
      /*c_mod=*/(short)0, c, /*reuse_a=*/false, /*reuse_b=*/false);
}

// ---------------------------------------------------------------------------
// Fragment loaders (layouts per CDNA5 ISA §7.12.2, wave32)
//   A (16x32 f16): lane L holds row M=L&15; half i -> K = 8*(L>=16) + 16*(i>=8) + (i&7)
//   B (32x16 f16): lane L holds col N=L&15; half i -> K = 8*(L>=16) + 16*(i>=8) + (i&7)
//   C/D (16x16 f32): lane L holds col N=L&15; vgpr r -> row M = r + 8*(L>=16)
// ---------------------------------------------------------------------------
DEV v16h load_a_f16(const _Float16* __restrict__ base, int row0, int ld,
                    int kbase, int lane) {
  int m  = lane & 15;
  int bk = (lane >> 4) ? 8 : 0;
  const _Float16* p = base + (size_t)(row0 + m) * ld + kbase + bk;
  v8h lo = *(const v8h*)(p);
  v8h hi = *(const v8h*)(p + 16);
  return join8(lo, hi);
}

DEV v16h load_a_f32cvt(const float* __restrict__ base, int row0, int ld,
                       int kbase, int lane) {
  int m  = lane & 15;
  int bk = (lane >> 4) ? 8 : 0;
  const float* p = base + (size_t)(row0 + m) * ld + kbase + bk;
  v16h r;
#pragma unroll
  for (int i = 0; i < 8; ++i) {
    r[i]     = (_Float16)p[i];
    r[8 + i] = (_Float16)p[i + 16];
  }
  return r;
}

// colT layout: element (k, n) stored at colT[n*ldk + k]  (column-contiguous)
DEV v16h load_b_colT(const _Float16* __restrict__ colT, int n0, int ldk,
                     int kbase, int lane) {
  int n  = n0 + (lane & 15);
  int bk = (lane >> 4) ? 8 : 0;
  const _Float16* p = colT + (size_t)n * ldk + kbase + bk;
  return join8(*(const v8h*)(p), *(const v8h*)(p + 16));
}

// ---------------------------------------------------------------------------
// Kernel 0: weight transpose + f32->f16  (out[c*R + r] = in[r*C + c])
// ---------------------------------------------------------------------------
__global__ void k_transpose_w(const float* __restrict__ in,
                              _Float16* __restrict__ out, int R, int C) {
  int i = blockIdx.x * blockDim.x + threadIdx.x;
  if (i >= R * C) return;
  int r = i / C, c = i % C;
  out[(size_t)c * R + r] = (_Float16)in[i];
}

// ---------------------------------------------------------------------------
// Kernel 1: projection GEMM  Y[M,64] = (X[M,128] @ W + bias) * scale
//   mode 0: Y row-major f16 [M][64]
//   mode 1: Y transposed per batch of 4096 rows: [b][64][4096] (for V)
// One wave computes a 16x64 tile. Grid is sized so every wave-tile is full.
// ---------------------------------------------------------------------------
__global__ void __launch_bounds__(256)
k_proj(const float* __restrict__ X, const _Float16* __restrict__ wT,
       const float* __restrict__ bias, _Float16* __restrict__ Yrm,
       _Float16* __restrict__ Ytr, int M, float scale, int mode) {
  int lane = threadIdx.x & 31;
  int wv   = threadIdx.x >> 5;
  int tile = blockIdx.x * (blockDim.x >> 5) + wv;
  if (tile * 16 >= M) return;
  int row0 = tile * 16;

  v8f acc[4];
#pragma unroll
  for (int nt = 0; nt < 4; ++nt) acc[nt] = zero8();

#pragma unroll
  for (int c = 0; c < 4; ++c) {                 // K = 128, steps of 32
    v16h a = load_a_f32cvt(X, row0, DIN_, c * 32, lane);
#pragma unroll
    for (int nt = 0; nt < 4; ++nt) {
      v16h b = load_b_colT(wT, nt * 16, DIN_, c * 32, lane);
      acc[nt] = wmma_f16(a, b, acc[nt]);
    }
  }

  int col = lane & 15;
  int rb  = (lane >> 4) ? 8 : 0;
#pragma unroll
  for (int nt = 0; nt < 4; ++nt) {
    int n   = nt * 16 + col;
    float bb = bias[n];
#pragma unroll
    for (int r = 0; r < 8; ++r) {
      float y = (acc[nt][r] + bb) * scale;
      int row = row0 + rb + r;
      if (mode == 0) {
        Yrm[(size_t)row * DK_ + n] = (_Float16)y;
      } else {
        // vT[b][n][key]: b = row/4096, key = row%4096
        Ytr[((size_t)(row >> 12) * DK_ + n) * NO_ + (row & (NO_ - 1))] =
            (_Float16)y;
      }
    }
  }
}

// ---------------------------------------------------------------------------
// Kernel 2: fused flash-style attention.
// One wave owns a 16-query tile; loops over 4096 keys in chunks of 32.
// qh is pre-scaled by 1/sqrt(DK). Online row-max/row-sum; diagonal + obj_mask
// applied to P (max is taken pre-mask, matching the reference). P transposed
// C-layout -> A-layout through per-wave LDS.
// ---------------------------------------------------------------------------
__global__ void __launch_bounds__(128)
k_attn(const _Float16* __restrict__ qh, const _Float16* __restrict__ kh,
       const _Float16* __restrict__ vTh, const unsigned char* __restrict__ om,
       _Float16* __restrict__ v_h) {
  __shared__ __align__(32) _Float16 Pbuf[4][16 * 32];

  int lane = threadIdx.x & 31;
  int wv   = threadIdx.x >> 5;
  int tile = blockIdx.x * 4 + wv;          // 0..511
  int b    = tile >> 6;                    // 64 query-tiles per batch
  int q0   = (tile & 63) * 16;             // query base within batch

  const _Float16* qb = qh + (size_t)(b * NC_ + q0) * DK_;
  const _Float16* kb = kh + (size_t)b * NO_ * DK_;
  const _Float16* vb = vTh + (size_t)b * DV_ * NO_;
  const unsigned char* mb = om + (size_t)b * NO_;
  _Float16* Pw = Pbuf[wv];

  // Q fragments: whole 16x64 tile (2 k-chunks), loaded once
  v16h aq0 = load_a_f16(qb, 0, DK_, 0, lane);
  v16h aq1 = load_a_f16(qb, 0, DK_, 32, lane);

  float m[8], l[8];
#pragma unroll
  for (int r = 0; r < 8; ++r) { m[r] = -1e30f; l[r] = 0.0f; }
  v8f o[4];
#pragma unroll
  for (int nt = 0; nt < 4; ++nt) o[nt] = zero8();

  int col = lane & 15;
  int rb  = (lane >> 4) ? 8 : 0;

  for (int kk = 0; kk < NO_; kk += 32) {
    // ---- S = Q @ K^T for 32 keys (two 16-col tiles) ----
    v8f s0 = zero8(), s1 = zero8();
    s0 = wmma_f16(aq0, load_b_colT(kb, kk, DK_, 0, lane), s0);
    s0 = wmma_f16(aq1, load_b_colT(kb, kk, DK_, 32, lane), s0);
    s1 = wmma_f16(aq0, load_b_colT(kb, kk + 16, DK_, 0, lane), s1);
    s1 = wmma_f16(aq1, load_b_colT(kb, kk + 16, DK_, 32, lane), s1);

    // ---- online max update (pre-mask, as in reference) ----
#pragma unroll
    for (int r = 0; r < 8; ++r) {
      float v = fmaxf(s0[r], s1[r]);
      v = fmaxf(v, __shfl_xor(v, 1));
      v = fmaxf(v, __shfl_xor(v, 2));
      v = fmaxf(v, __shfl_xor(v, 4));
      v = fmaxf(v, __shfl_xor(v, 8));
      float nm = fmaxf(m[r], v);
      float sc = __expf(m[r] - nm);
      l[r] *= sc;
#pragma unroll
      for (int nt = 0; nt < 4; ++nt) o[nt][r] *= sc;
      m[r] = nm;
    }

    // ---- P = exp(S - m), mask diag + obj_mask, row-sum, store to LDS ----
    int key0 = kk + col, key1 = kk + 16 + col;
    unsigned char mk0 = mb[key0], mk1 = mb[key1];
#pragma unroll
    for (int r = 0; r < 8; ++r) {
      int qg = q0 + rb + r;                 // query index within batch
      float p0 = __expf(s0[r] - m[r]);
      float p1 = __expf(s1[r] - m[r]);
      if (!mk0 || key0 == qg) p0 = 0.0f;
      if (!mk1 || key1 == qg) p1 = 0.0f;
      Pw[(rb + r) * 32 + col]      = (_Float16)p0;
      Pw[(rb + r) * 32 + 16 + col] = (_Float16)p1;
      float rs = p0 + p1;
      rs += __shfl_xor(rs, 1);
      rs += __shfl_xor(rs, 2);
      rs += __shfl_xor(rs, 4);
      rs += __shfl_xor(rs, 8);
      l[r] += rs;
    }

    asm volatile("s_wait_dscnt 0x0" ::: "memory");

    // ---- O += P(16x32) @ V(32x64) ----
    {
      int bk = (lane >> 4) ? 8 : 0;
      const _Float16* pp = Pw + col * 32 + bk;   // A-layout: row = lane&15
      v16h ap = join8(*(const v8h*)pp, *(const v8h*)(pp + 16));
#pragma unroll
      for (int nt = 0; nt < 4; ++nt) {
        v16h bv = load_b_colT(vb, nt * 16, NO_, kk, lane);
        o[nt] = wmma_f16(ap, bv, o[nt]);
      }
    }
    asm volatile("s_wait_dscnt 0x0" ::: "memory");
  }

  // ---- epilogue: v_ = O / (l + 1e-6), store f16 ----
#pragma unroll
  for (int nt = 0; nt < 4; ++nt) {
    int n = nt * 16 + col;
#pragma unroll
    for (int r = 0; r < 8; ++r) {
      float y = o[nt][r] / (l[r] + 1e-6f);
      int row = b * NC_ + q0 + rb + r;
      v_h[(size_t)row * DV_ + n] = (_Float16)y;
    }
  }
}

// ---------------------------------------------------------------------------
// Kernel 3: fused 2-layer MLP heads.
// x = [v_, le] (192) per 16-row tile; layer1 -> LeakyReLU -> LDS (f16),
// layer2 -> f32 out[row][head*32 + n].
// ---------------------------------------------------------------------------
__global__ void __launch_bounds__(128)
k_heads(const _Float16* __restrict__ v_h, const float* __restrict__ le,
        const _Float16* __restrict__ w1mT, const float* __restrict__ b1m,
        const _Float16* __restrict__ w2mT, const float* __restrict__ b2m,
        const _Float16* __restrict__ w1sT, const float* __restrict__ b1s,
        const _Float16* __restrict__ w2sT, const float* __restrict__ b2s,
        float* __restrict__ out) {
  __shared__ __align__(32) _Float16 H[4][16 * DH_];   // 32 KB

  int lane = threadIdx.x & 31;
  int wv   = threadIdx.x >> 5;
  int tile = blockIdx.x * 4 + wv;          // 0..511
  int row0 = tile * 16;
  _Float16* Hw = H[wv];

  // x fragments: K = 192 = v_(64) ++ le(128), six 32-chunks
  v16h ax[6];
  ax[0] = load_a_f16(v_h, row0, DV_, 0, lane);
  ax[1] = load_a_f16(v_h, row0, DV_, 32, lane);
#pragma unroll
  for (int c = 0; c < 4; ++c)
    ax[2 + c] = load_a_f32cvt(le, row0, DIN_, c * 32, lane);

  int col = lane & 15;
  int rb  = (lane >> 4) ? 8 : 0;

#pragma unroll
  for (int head = 0; head < 2; ++head) {
    const _Float16* w1T = head ? w1sT : w1mT;
    const float*    b1  = head ? b1s  : b1m;
    const _Float16* w2T = head ? w2sT : w2mT;
    const float*    b2  = head ? b2s  : b2m;

    // ---- layer 1: h = leaky(x @ W1 + b1), 16 col-tiles of DH=256 ----
#pragma unroll 4
    for (int nt = 0; nt < 16; ++nt) {
      v8f acc = zero8();
#pragma unroll
      for (int c = 0; c < 6; ++c)
        acc = wmma_f16(ax[c], load_b_colT(w1T, nt * 16, DIN_ + DV_, c * 32, lane), acc);
      int n   = nt * 16 + col;
      float bb = b1[n];
#pragma unroll
      for (int r = 0; r < 8; ++r) {
        float h = acc[r] + bb;
        h = (h > 0.0f) ? h : 0.01f * h;
        Hw[(rb + r) * DH_ + n] = (_Float16)h;
      }
    }

    asm volatile("s_wait_dscnt 0x0" ::: "memory");

    // ---- layer 2: y = h @ W2 + b2, 2 col-tiles of DP=32 ----
#pragma unroll
    for (int nt2 = 0; nt2 < 2; ++nt2) {
      v8f a2 = zero8();
#pragma unroll
      for (int c2 = 0; c2 < 8; ++c2) {
        int bk = (lane >> 4) ? 8 : 0;
        const _Float16* pp = Hw + col * DH_ + c2 * 32 + bk;
        v16h ah = join8(*(const v8h*)pp, *(const v8h*)(pp + 16));
        a2 = wmma_f16(ah, load_b_colT(w2T, nt2 * 16, DH_, c2 * 32, lane), a2);
      }
      int n   = nt2 * 16 + col;
      float bb = b2[n];
#pragma unroll
      for (int r = 0; r < 8; ++r) {
        int row = row0 + rb + r;
        out[(size_t)row * (2 * DP_) + head * DP_ + n] = a2[r] + bb;
      }
    }
    asm volatile("s_wait_dscnt 0x0" ::: "memory");
  }
}

// ---------------------------------------------------------------------------
// Host launcher
// ---------------------------------------------------------------------------
extern "C" void kernel_launch(void* const* d_in, const int* in_sizes, int n_in,
                              void* d_out, int out_size, void* d_ws,
                              size_t ws_size, hipStream_t stream) {
  const float* le = (const float*)d_in[0];   // [8,1024,128]
  const float* g  = (const float*)d_in[1];   // [8,4096,128]
  const unsigned char* om = (const unsigned char*)d_in[2];  // bool [8,4096]
  const float* Wq = (const float*)d_in[3];  const float* bq = (const float*)d_in[4];
  const float* Wk = (const float*)d_in[5];  const float* bk = (const float*)d_in[6];
  const float* Wv = (const float*)d_in[7];  const float* bv = (const float*)d_in[8];
  const float* W1m = (const float*)d_in[9];  const float* b1m = (const float*)d_in[10];
  const float* W2m = (const float*)d_in[11]; const float* b2m = (const float*)d_in[12];
  const float* W1s = (const float*)d_in[13]; const float* b1s = (const float*)d_in[14];
  const float* W2s = (const float*)d_in[15]; const float* b2s = (const float*)d_in[16];
  float* out = (float*)d_out;                // [8,1024,64]

  // Workspace layout (bytes)
  char* ws = (char*)d_ws;
  _Float16* qh   = (_Float16*)(ws + 0);                   // 8192*64   f16
  _Float16* kh   = (_Float16*)(ws + 1048576);             // 32768*64  f16
  _Float16* vTh  = (_Float16*)(ws + 5242880);             // [8][64][4096] f16
  _Float16* v_h  = (_Float16*)(ws + 9437184);             // 8192*64   f16
  _Float16* wqT  = (_Float16*)(ws + 10485760);            // [64][128]
  _Float16* wkT  = (_Float16*)(ws + 10502144);            // [64][128]
  _Float16* wvT  = (_Float16*)(ws + 10518528);            // [64][128]
  _Float16* w1mT = (_Float16*)(ws + 10534912);            // [256][192]
  _Float16* w2mT = (_Float16*)(ws + 10633216);            // [32][256]
  _Float16* w1sT = (_Float16*)(ws + 10649600);            // [256][192]
  _Float16* w2sT = (_Float16*)(ws + 10747904);            // [32][256]

  auto tw = [&](const float* w, _Float16* o, int R, int C) {
    int n = R * C;
    k_transpose_w<<<(n + 255) / 256, 256, 0, stream>>>(w, o, R, C);
  };
  tw(Wq, wqT, DIN_, DK_);
  tw(Wk, wkT, DIN_, DK_);
  tw(Wv, wvT, DIN_, DK_);
  tw(W1m, w1mT, DIN_ + DV_, DH_);
  tw(W2m, w2mT, DH_, DP_);
  tw(W1s, w1sT, DIN_ + DV_, DH_);
  tw(W2s, w2sT, DH_, DP_);

  const float inv_sqrt_dk = 0.125f;  // 1/sqrt(64)
  // q: 8192 rows -> 512 tiles -> 64 blocks of 8 waves
  k_proj<<<64, 256, 0, stream>>>(le, wqT, bq, qh, (_Float16*)nullptr,
                                 MQ_, inv_sqrt_dk, 0);
  // k: 32768 rows -> 2048 tiles -> 256 blocks
  k_proj<<<256, 256, 0, stream>>>(g, wkT, bk, kh, (_Float16*)nullptr,
                                  MG_, 1.0f, 0);
  // v: transposed output
  k_proj<<<256, 256, 0, stream>>>(g, wvT, bv, (_Float16*)nullptr, vTh,
                                  MG_, 1.0f, 1);

  // attention: 512 query tiles, 4 waves/block
  k_attn<<<128, 128, 0, stream>>>(qh, kh, vTh, om, v_h);

  // heads: 512 row tiles, 4 waves/block
  k_heads<<<128, 128, 0, stream>>>(v_h, le, w1mT, b1m, w2mT, b2m,
                                   w1sT, b1s, w2sT, b2s, out);
}
